// GraphAttentionLayer_73212012528271
// MI455X (gfx1250) — matrix operand order repaired and checked by reading, not compiled, not run on hardware
//
#include <hip/hip_runtime.h>
#include <hip/hip_bf16.h>
#include <math.h>

// CDNA5 / gfx1250: wave32, WMMA 16x16x32 bf16 (f32 accumulate),
// TDM tensor_load_to_lds staging (TENSORcnt), global_prefetch_b8.
typedef __attribute__((ext_vector_type(16))) __bf16 v16bf;
typedef __attribute__((ext_vector_type(8)))  float  v8f;
typedef __attribute__((ext_vector_type(4)))  unsigned int u32x4;
typedef __attribute__((ext_vector_type(8)))  unsigned int u32x8;

#define F_IN    128
#define F_OUT   64
#define NNODES  4096
#define NBATCH  2
#define NROWS   (NBATCH * NNODES)   // 8192 flattened rows
#define ALPHA   0.2f
#define MINV    1e-15f

#define USE_TDM 1   // 1: one tensor_load_to_lds per tile; 0: per-thread async b128

// ---------------------------------------------------------------------------
// Kernel 1: h = X @ W   ([8192,128] x [128,64], bf16 WMMA, f32 accum)
// Writes h (f32 row-major, for the s1/s2 kernel) and hbfT (bf16 COLUMN-major
// [b][o][j]) so the attention kernel can stage H tiles as rectangular 2D
// DMA copies (TDM) with no transpose at consume time.
// ---------------------------------------------------------------------------
__global__ void __launch_bounds__(128)
gat_hw_kernel(const float* __restrict__ X, const float* __restrict__ W,
              float* __restrict__ h, __bf16* __restrict__ hbfT)
{
    __shared__ __bf16 wT[F_OUT * F_IN];       // 16 KB, W transposed (col-major)
    const int tid  = threadIdx.x;
    const int wave = tid >> 5;
    const int lane = tid & 31;
    const int lr   = lane & 15;               // row-in-A-tile / col-in-B-tile
    const int hi   = lane >> 4;               // K-group select

    for (int idx = tid; idx < F_OUT * F_IN; idx += 128) {
        const int k = idx >> 6, o = idx & 63;
        wT[o * F_IN + k] = (__bf16)W[idx];
    }
    __syncthreads();

    const int rowbase = blockIdx.x * 64 + wave * 16;
    const int arow    = rowbase + lr;

    v8f acc0 = {0,0,0,0,0,0,0,0};
    v8f acc1 = acc0, acc2 = acc0, acc3 = acc0;

    #pragma unroll
    for (int kb = 0; kb < 4; ++kb) {
        v16bf a;
        const float* xr = X + (size_t)arow * F_IN + kb * 32 + hi * 16;
        #pragma unroll
        for (int i = 0; i < 8; ++i) {
            a[i]     = (__bf16)xr[i];          // K = hi*16 + i
            a[i + 8] = (__bf16)xr[8 + i];      // K = hi*16 + 8 + i
        }
        const int koff = kb * 32 + hi * 16;
        #pragma unroll
        for (int t = 0; t < 4; ++t) {
            const v16bf bv = *(const v16bf*)&wT[(t * 16 + lr) * F_IN + koff];
            v8f& acc = (t == 0) ? acc0 : (t == 1) ? acc1 : (t == 2) ? acc2 : acc3;
            acc = __builtin_amdgcn_wmma_f32_16x16x32_bf16(
                      false, a, false, bv, (short)0, acc, false, false);
        }
    }

    // C layout: lane holds col N = lane&15, rows M = v + 8*(lane>>4)
    #pragma unroll
    for (int t = 0; t < 4; ++t) {
        const v8f acc = (t == 0) ? acc0 : (t == 1) ? acc1 : (t == 2) ? acc2 : acc3;
        #pragma unroll
        for (int v = 0; v < 8; ++v) {
            const int grow = rowbase + v + 8 * hi;
            const int col  = t * 16 + lr;
            const float val = acc[v];
            h[(size_t)grow * F_OUT + col] = val;
            const int bb = grow >> 12, jl = grow & (NNODES - 1);
            hbfT[((size_t)bb * F_OUT + col) * NNODES + jl] = (__bf16)val;
        }
    }
}

// ---------------------------------------------------------------------------
// Kernel 1b: s1 = h @ a1, s2 = h @ a2  (one thread per row; trivial)
// ---------------------------------------------------------------------------
__global__ void __launch_bounds__(256)
gat_s_kernel(const float* __restrict__ h, const float* __restrict__ a,
             float* __restrict__ s1, float* __restrict__ s2)
{
    const int row = blockIdx.x * 256 + threadIdx.x;
    if (row >= NROWS) return;
    const float* hr = h + (size_t)row * F_OUT;
    float r1 = 0.f, r2 = 0.f;
    #pragma unroll 8
    for (int o = 0; o < F_OUT; ++o) {
        const float v = hr[o];
        r1 = fmaf(v, a[o], r1);
        r2 = fmaf(v, a[F_OUT + o], r2);
    }
    s1[row] = r1;
    s2[row] = r2;
}

// ---------------------------------------------------------------------------
// Kernel 2: fused masked softmax attention + P@H + ELU (flash-style).
// 128 threads = 4 waves per block; block owns 64 softmax rows of one batch.
// adj streams straight to VGPRs (the 134 MB / 5.8 us bandwidth floor); the
// H tile (64x64 bf16 of h^T) is double-buffered in LDS via ONE TDM
// tensor_load_to_lds per tile (wave-0 issued, TENSORcnt-tracked), so the
// DMA of block jb+1 fully overlaps the softmax+WMMA work on block jb.
// ---------------------------------------------------------------------------
__global__ void __launch_bounds__(128)
gat_attn_kernel(const int* __restrict__ adj, const __bf16* __restrict__ hbfT,
                const float* __restrict__ s1, const float* __restrict__ s2,
                float* __restrict__ out)
{
    __shared__ __align__(64) __bf16 hT[2][F_OUT * 64];  // 2 x 8 KB double buffer
    const int tid  = threadIdx.x;
    const int wave = tid >> 5;
    const int lane = tid & 31;
    const int lr   = lane & 15;
    const int hi   = lane >> 4;

    const int rowbase = blockIdx.x * 64 + wave * 16;   // flattened row base
    const int b       = (blockIdx.x * 64) >> 12;       // 4096 rows per batch
    const int myrow   = rowbase + lr;                  // this lane's softmax row
    const int irow    = myrow & (NNODES - 1);
    const long adjbase = (long)b * NNODES * NNODES + (long)irow * NNODES;

    const float   s1v = s1[myrow];
    const float*  s2b = s2 + (size_t)b * NNODES;
    const char*   hTg = (const char*)(hbfT + (size_t)b * F_OUT * NNODES);

#if USE_TDM
    // Uniform (scalar-branch) predicate: only wave 0 issues/awaits TDM ops.
    const bool issuer = (__builtin_amdgcn_readfirstlane(wave) == 0);
    auto tdm_stage = [&](int jb, int bufsel) {
        const unsigned long long ga = (unsigned long long)(hTg + (size_t)jb * 128);
        const unsigned la = (unsigned)(unsigned long long)&hT[bufsel][0];
        u32x4 g0;
        g0.x = 1u;                                  // count=1 (valid user D#)
        g0.y = la;                                  // lds_addr (bytes)
        g0.z = (unsigned)ga;                        // global_addr[31:0]
        g0.w = (unsigned)(ga >> 32) | (2u << 30);   // global_addr[56:32] | type=2
        u32x8 g1;
        g1[0] = 1u << 16;                           // data_size=1 -> 2-byte elems
        g1[1] = ((unsigned)NNODES & 0xffffu) << 16; // tensor_dim0[15:0] = 4096
        g1[2] = ((unsigned)NNODES >> 16) | (64u << 16); // dim0[31:16] | tensor_dim1=64
        g1[3] = 64u << 16;                          // dim1[31:16]=0 | tile_dim0=64
        g1[4] = 64u;                                // tile_dim1=64, tile_dim2=0
        g1[5] = (unsigned)NNODES;                   // tensor_dim0_stride[31:0]=4096
        g1[6] = 0u;                                 // stride[47:32] | dim1_stride lo
        g1[7] = 0u;
        const u32x4 g2 = {0u, 0u, 0u, 0u};          // unused dims (2D tensor)
        const u32x4 g3 = {0u, 0u, 0u, 0u};
        asm volatile("tensor_load_to_lds %0, %1, %2, %3"
                     :: "s"(g0), "s"(g1), "s"(g2), "s"(g3) : "memory");
    };
#else
    auto stage = [&](int jb, int bufsel) {
        const unsigned lbase = (unsigned)(unsigned long long)&hT[bufsel][0];
        #pragma unroll
        for (int k = 0; k < 4; ++k) {
            const int c = tid * 4 + k;              // chunk id 0..511 (16B each)
            const int o = c >> 3, part = c & 7;
            const unsigned long long ga =
                (unsigned long long)(hTg + (size_t)o * (NNODES * 2)
                                         + (size_t)jb * 128 + part * 16);
            const unsigned la = lbase + (unsigned)(o * 128 + part * 16);
            asm volatile("global_load_async_to_lds_b128 %0, %1, off"
                         :: "v"(la), "v"(ga) : "memory");
        }
    };
#endif

    float mrun = -3.0e38f, lrun = 0.0f;
    v8f acc0 = {0,0,0,0,0,0,0,0};
    v8f acc1 = acc0, acc2 = acc0, acc3 = acc0;

#if USE_TDM
    if (issuer) {
        tdm_stage(0, 0);
        __builtin_amdgcn_s_wait_tensorcnt(0);
    }
#else
    stage(0, 0);
    asm volatile("s_wait_asynccnt 0x0" ::: "memory");
#endif
    __syncthreads();

    for (int jb = 0; jb < NNODES / 64; ++jb) {
        const int cur = jb & 1;
#if USE_TDM
        if (issuer && (jb + 1 < NNODES / 64)) tdm_stage(jb + 1, cur ^ 1);
#else
        if (jb + 1 < NNODES / 64) stage(jb + 1, cur ^ 1);
#endif

        #pragma unroll
        for (int kc = 0; kc < 2; ++kc) {
            // ---- 16 attention logits per lane, WMMA A-layout K order -------
            const int c0 = jb * 64 + kc * 32 + hi * 16;
            __builtin_prefetch(&adj[adjbase + c0 + 64], 0, 1); // global_prefetch_b8
            float e[16];
            #pragma unroll
            for (int g = 0; g < 2; ++g) {
                const int cg = c0 + g * 8;     // K = hi*16 + g*8 + 0..7
                const int4*   qa = (const int4*)&adj[adjbase + cg];
                const float4* qs = (const float4*)&s2b[cg];
                const int4   a0 = qa[0], a1 = qa[1];
                const float4 f0 = qs[0], f1 = qs[1];
                const int   ai[8] = {a0.x, a0.y, a0.z, a0.w, a1.x, a1.y, a1.z, a1.w};
                const float sv[8] = {f0.x, f0.y, f0.z, f0.w, f1.x, f1.y, f1.z, f1.w};
                #pragma unroll
                for (int i = 0; i < 8; ++i) {
                    const float raw = s1v + sv[i];
                    const float lk  = raw > 0.f ? raw : ALPHA * raw;
                    e[g * 8 + i] = (ai[i] > 0) ? lk : MINV;
                }
            }

            // ---- online softmax update (wave32: one SWAPX16 combine) -------
            float mloc = e[0];
            #pragma unroll
            for (int k = 1; k < 16; ++k) mloc = fmaxf(mloc, e[k]);
            mloc = fmaxf(mloc, __shfl_xor(mloc, 16, 32));
            const float mnew  = fmaxf(mrun, mloc);
            const float scale = __expf(mrun - mnew);

            float psum = 0.f;
            v16bf ap;
            #pragma unroll
            for (int k = 0; k < 16; ++k) {
                const float p = __expf(e[k] - mnew);
                psum += p;
                ap[k] = (__bf16)p;
            }
            psum += __shfl_xor(psum, 16, 32);
            lrun = lrun * scale + psum;
            mrun = mnew;

            // rescale accumulators: gather per-row scale onto C-layout rows
            float sc[8];
            #pragma unroll
            for (int v = 0; v < 8; ++v) sc[v] = __shfl(scale, v + 8 * hi, 32);
            #pragma unroll
            for (int v = 0; v < 8; ++v) {
                acc0[v] *= sc[v]; acc1[v] *= sc[v];
                acc2[v] *= sc[v]; acc3[v] *= sc[v];
            }

            // ---- P(16x32) @ H(32x64) via 4 bf16 WMMAs ----------------------
            const int koff = kc * 32 + hi * 16;
            #pragma unroll
            for (int t = 0; t < 4; ++t) {
                const v16bf bv = *(const v16bf*)&hT[cur][(t * 16 + lr) * 64 + koff];
                v8f& acc = (t == 0) ? acc0 : (t == 1) ? acc1 : (t == 2) ? acc2 : acc3;
                acc = __builtin_amdgcn_wmma_f32_16x16x32_bf16(
                          false, ap, false, bv, (short)0, acc, false, false);
            }
        }

        // next buffer's DMA must land before it becomes current
#if USE_TDM
        if (issuer) __builtin_amdgcn_s_wait_tensorcnt(0);
#else
        asm volatile("s_wait_asynccnt 0x0" ::: "memory");
#endif
        __syncthreads();
    }

    // ---- finalize: divide by l, ELU, store --------------------------------
    float linv[8];
    #pragma unroll
    for (int v = 0; v < 8; ++v)
        linv[v] = 1.0f / __shfl(lrun, v + 8 * hi, 32);

    #pragma unroll
    for (int t = 0; t < 4; ++t) {
        const v8f acc = (t == 0) ? acc0 : (t == 1) ? acc1 : (t == 2) ? acc2 : acc3;
        #pragma unroll
        for (int v = 0; v < 8; ++v) {
            const float x = acc[v] * linv[v];
            const float r = x > 0.f ? x : (__expf(x) - 1.0f);
            out[(size_t)(rowbase + v + 8 * hi) * F_OUT + t * 16 + lr] = r;
        }
    }
}

// ---------------------------------------------------------------------------
extern "C" void kernel_launch(void* const* d_in, const int* in_sizes, int n_in,
                              void* d_out, int out_size, void* d_ws, size_t ws_size,
                              hipStream_t stream)
{
    (void)in_sizes; (void)n_in; (void)out_size; (void)ws_size;
    const float* X   = (const float*)d_in[0];   // [2,4096,128] f32
    const int*   adj = (const int*)d_in[1];     // [2,4096,4096] i32
    const float* W   = (const float*)d_in[2];   // [128,64] f32
    const float* a   = (const float*)d_in[3];   // [128,1] f32
    float* out = (float*)d_out;                 // [2,4096,64] f32

    char* ws = (char*)d_ws;
    float*  h    = (float*)ws;                                  // 2 MB
    __bf16* hbfT = (__bf16*)(ws + (size_t)NROWS * F_OUT * 4);   // 1 MB, [b][o][j]
    float*  s1   = (float*)(ws + (size_t)NROWS * F_OUT * 6);    // 32 KB
    float*  s2   = (float*)(ws + (size_t)NROWS * F_OUT * 6 + (size_t)NROWS * 4);

    gat_hw_kernel  <<<NROWS / 64, 128, 0, stream>>>(X, W, h, hbfT);
    gat_s_kernel   <<<NROWS / 256, 256, 0, stream>>>(h, a, s1, s2);
    gat_attn_kernel<<<NROWS / 64, 128, 0, stream>>>(adj, hbfT, s1, s2, out);
}
